// BaseAttender_37855841746965
// MI455X (gfx1250) — matrix operand order repaired and checked by reading, not compiled, not run on hardware
//
#include <hip/hip_runtime.h>
#include <hip/hip_fp16.h>

// ---------------------------------------------------------------------------
// Problem constants (from reference setup_inputs):
//   B=8, Q=2048, K=2048, D=512, DV=512, DO=1024
// d_out = [out (B*Q*DO f32)] ++ [attn (B*Q*K f32)]
// d_ws  = out1 (B*Q*DV f32) scratch
// ---------------------------------------------------------------------------

typedef _Float16 f16x16 __attribute__((ext_vector_type(16)));
typedef float    f32x8  __attribute__((ext_vector_type(8)));

// pack two f32 -> two f16 in one dword: v_cvt_pk_rtz_f16_f32 (1 VALU / 2 elems)
__device__ __forceinline__ unsigned pk2(float lo, float hi) {
    return __builtin_bit_cast(unsigned, __builtin_amdgcn_cvt_pkrtz(lo, hi));
}

__device__ __forceinline__ f16x16 ld_frag32B(const unsigned short* p0,
                                             const unsigned short* p1) {
    union { f16x16 v; uint4 q[2]; } u;
    u.q[0] = *(const uint4*)p0;   // 16B-aligned LDS read -> ds_load_b128
    u.q[1] = *(const uint4*)p1;
    return u.v;
}

__device__ __forceinline__ void cvt8(unsigned short* d, float4 x, float4 y) {
    unsigned* d32 = (unsigned*)d;          // 4B-aligned (offsets are 16B multiples)
    d32[0] = pk2(x.x, x.y);
    d32[1] = pk2(x.z, x.w);
    d32[2] = pk2(y.x, y.y);
    d32[3] = pk2(y.z, y.w);
}

// ---------------------------------------------------------------------------
// f16-staged f32 GEMM:  C[m,n] = alpha * sum_k A[m,k] * B(k,n)  (+ bias[n])
//   TRANSB=false: B stored [n][k] row-major (ldb = k-stride)  -> QK^T case
//   TRANSB=true : B stored [k][n] row-major (ldb = n-stride), staged transposed
// Block: 256 threads (8 waves), 128x128 C tile, K-step 32, 2-stage LDS double
// buffer (one barrier per k-step; next tile's global loads fly under WMMAs).
// Wave w: rows (w&3)*32 (2 m-subtiles), cols (w>>2)*64 (4 n-subtiles)
// => 8 WMMAs per wave per k-step; 64 WMMAs per block per 32KB staged.
// ---------------------------------------------------------------------------
template <bool TRANSB, bool HAS_BIAS>
__global__ __launch_bounds__(256)
void gemm_f16_wmma(const float* __restrict__ A, const float* __restrict__ Bm,
                   const float* __restrict__ bias, float* __restrict__ C,
                   int Kd, int lda, int ldb, int ldc,
                   long strideA, long strideB, long strideC, float alpha) {
    const int bz = blockIdx.z;
    A += bz * strideA;  Bm += bz * strideB;  C += bz * strideC;

    const int m0 = blockIdx.y * 128;
    const int n0 = blockIdx.x * 128;
    const int tid  = threadIdx.x;
    const int wave = tid >> 5;
    const int lane = tid & 31;

    __shared__ unsigned short As[2][128][40];   // 2 x 128x32 f16, 80B rows
    __shared__ unsigned short Bs[2][128][40];   // Bs[b][n][k]

    f32x8 acc[2][4];
    #pragma unroll
    for (int mt = 0; mt < 2; ++mt)
        #pragma unroll
        for (int nt = 0; nt < 4; ++nt)
            acc[mt][nt] = (f32x8){0.f,0.f,0.f,0.f,0.f,0.f,0.f,0.f};

    const int wmrow = (wave & 3) * 32;      // wave's row base within tile
    const int wncol = (wave >> 2) * 64;     // wave's col base within tile

    const int lrow  = lane & 15;
    const int abase = (lane >> 4) * 8;      // A frag: K {0..7,16..23}/{8..15,24..31}
    const int bbase = (lane >> 4) * 8;      // B frag: K {0..7,16..23}/{8..15,24..31}

    // staging coordinates (each thread moves 16 f32 per tile)
    const int sr = tid >> 1;                // 0..127
    const int sc = (tid & 1) * 16;          // 0 or 16
    const int tk = tid >> 3;                // 0..31   (TRANSB)
    const int tn = (tid & 7) * 16;          // 0..112  (TRANSB)

    float4 ax[4], bx[4];

    auto ldA = [&](int k0) {
        const float4* gp = (const float4*)(A + (long)(m0 + sr) * lda + k0 + sc);
        __builtin_prefetch(A + (long)(m0 + sr) * lda + k0 + 64 + sc, 0, 3);
        ax[0] = gp[0]; ax[1] = gp[1]; ax[2] = gp[2]; ax[3] = gp[3];
    };
    auto stA = [&](int b) {
        cvt8(&As[b][sr][sc],     ax[0], ax[1]);
        cvt8(&As[b][sr][sc + 8], ax[2], ax[3]);
    };
    auto ldB = [&](int k0) {
        if (!TRANSB) {
            const float4* gp = (const float4*)(Bm + (long)(n0 + sr) * ldb + k0 + sc);
            __builtin_prefetch(Bm + (long)(n0 + sr) * ldb + k0 + 64 + sc, 0, 3);
            bx[0] = gp[0]; bx[1] = gp[1]; bx[2] = gp[2]; bx[3] = gp[3];
        } else {
            const float4* gp = (const float4*)(Bm + (long)(k0 + tk) * ldb + n0 + tn);
            __builtin_prefetch(Bm + (long)(k0 + 64 + tk) * ldb + n0 + tn, 0, 3);
            bx[0] = gp[0]; bx[1] = gp[1]; bx[2] = gp[2]; bx[3] = gp[3];
        }
    };
    auto stB = [&](int b) {
        if (!TRANSB) {
            cvt8(&Bs[b][sr][sc],     bx[0], bx[1]);
            cvt8(&Bs[b][sr][sc + 8], bx[2], bx[3]);
        } else {
            const float v[16] = {bx[0].x,bx[0].y,bx[0].z,bx[0].w,
                                 bx[1].x,bx[1].y,bx[1].z,bx[1].w,
                                 bx[2].x,bx[2].y,bx[2].z,bx[2].w,
                                 bx[3].x,bx[3].y,bx[3].z,bx[3].w};
            #pragma unroll
            for (int i = 0; i < 8; ++i) {
                const unsigned p = pk2(v[2 * i], v[2 * i + 1]);
                Bs[b][tn + 2 * i][tk]     = (unsigned short)(p & 0xFFFFu);
                Bs[b][tn + 2 * i + 1][tk] = (unsigned short)(p >> 16);
            }
        }
    };

    // prologue: fill buffer 0
    ldA(0); ldB(0);
    stA(0); stB(0);
    __syncthreads();

    const int nsteps = Kd >> 5;
    for (int t = 0; t < nsteps; ++t) {
        const int cur = t & 1;
        const bool more = (t + 1) < nsteps;
        if (more) { ldA((t + 1) << 5); ldB((t + 1) << 5); }  // in flight under WMMAs

        f16x16 af[2], bfv[4];
        #pragma unroll
        for (int mt = 0; mt < 2; ++mt) {
            const int r = wmrow + mt * 16 + lrow;
            af[mt] = ld_frag32B(&As[cur][r][abase], &As[cur][r][abase + 16]);
        }
        #pragma unroll
        for (int nt = 0; nt < 4; ++nt) {
            const int r = wncol + nt * 16 + lrow;
            bfv[nt] = ld_frag32B(&Bs[cur][r][bbase], &Bs[cur][r][bbase + 16]);
        }
        #pragma unroll
        for (int mt = 0; mt < 2; ++mt)
            #pragma unroll
            for (int nt = 0; nt < 4; ++nt)
                acc[mt][nt] = __builtin_amdgcn_wmma_f32_16x16x32_f16(
                    false, af[mt], false, bfv[nt], (short)0, acc[mt][nt],
                    false, false);

        if (more) { stA(cur ^ 1); stB(cur ^ 1); }
        __syncthreads();
    }

    // ---- epilogue: C/D layout per ISA 7.12.2 (32-bit C 16x16) ----
    const int ncol  = lane & 15;
    const int rbump = (lane >> 4) * 8;
    #pragma unroll
    for (int nt = 0; nt < 4; ++nt) {
        const int col = n0 + wncol + nt * 16 + ncol;
        const float bb = HAS_BIAS ? bias[col] : 0.f;
        #pragma unroll
        for (int mt = 0; mt < 2; ++mt) {
            const long rowbase = (long)(m0 + wmrow + mt * 16 + rbump);
            #pragma unroll
            for (int i = 0; i < 8; ++i)
                C[(rowbase + i) * ldc + col] = acc[mt][nt][i] * alpha + bb;
        }
    }
}

// ---------------------------------------------------------------------------
// Sparsemax over one row of 2048 logits per 256-thread block (in place).
// ---------------------------------------------------------------------------
__global__ __launch_bounds__(256)
void sparsemax_rows(float* __restrict__ attn) {
    constexpr int N = 2048, T = 256, CH = N / T;   // 8 elems/thread
    __shared__ float s[N];
    __shared__ float aux[T];
    __shared__ int   kz_sh;
    __shared__ float tau_sh;

    float* row = attn + (long)blockIdx.x * N;
    const int tid = threadIdx.x;

    float zloc[CH];
    #pragma unroll
    for (int i = 0; i < CH; ++i) {          // coalesced load, keep originals
        int j = tid + i * T;
        float v = row[j];
        zloc[i] = v;
        s[j] = v;
    }
    __syncthreads();

    // bitonic sort, descending
    for (unsigned k = 2; k <= (unsigned)N; k <<= 1) {
        for (unsigned j = k >> 1; j > 0; j >>= 1) {
            for (unsigned i = tid; i < (unsigned)N; i += T) {
                unsigned ixj = i ^ j;
                if (ixj > i) {
                    float a = s[i], b = s[ixj];
                    bool desc = ((i & k) == 0);
                    if ((a < b) == desc) { s[i] = b; s[ixj] = a; }
                }
            }
            __syncthreads();
        }
    }

    if (tid == 0) kz_sh = 0;

    // blocked inclusive cumsum
    float v[CH], p[CH], run = 0.f;
    const int base = tid * CH;
    #pragma unroll
    for (int i = 0; i < CH; ++i) { v[i] = s[base + i]; run += v[i]; p[i] = run; }
    aux[tid] = run;
    __syncthreads();
    for (int off = 1; off < T; off <<= 1) {
        float x = aux[tid];
        float add = (tid >= off) ? aux[tid - off] : 0.f;
        __syncthreads();
        aux[tid] = x + add;
        __syncthreads();
    }
    const float excl = (tid > 0) ? aux[tid - 1] : 0.f;

    int cnt = 0;
    #pragma unroll
    for (int i = 0; i < CH; ++i) {
        const int gi = base + i;
        const float cs = excl + p[i];
        if (1.0f + (float)(gi + 1) * v[i] > cs) ++cnt;
        s[gi] = cs;                          // overwrite sorted with cumsum
    }
    atomicAdd(&kz_sh, cnt);
    __syncthreads();
    if (tid == 0) {
        const int kz = kz_sh;
        tau_sh = (s[kz - 1] - 1.0f) / (float)kz;
    }
    __syncthreads();
    const float tau = tau_sh;

    #pragma unroll
    for (int i = 0; i < CH; ++i) {
        int j = tid + i * T;
        row[j] = fmaxf(zloc[i] - tau, 0.0f);
    }
}

// ---------------------------------------------------------------------------
extern "C" void kernel_launch(void* const* d_in, const int* in_sizes, int n_in,
                              void* d_out, int out_size, void* d_ws, size_t ws_size,
                              hipStream_t stream) {
    (void)in_sizes; (void)n_in; (void)out_size; (void)ws_size;
    constexpr int B = 8, Q = 2048, KK = 2048, D = 512, DV = 512, DO = 1024;

    const float* keys    = (const float*)d_in[0];   // [B,K,D]
    const float* queries = (const float*)d_in[1];   // [B,Q,D]
    const float* values  = (const float*)d_in[2];   // [B,K,DV]
    const float* W       = (const float*)d_in[3];   // [DV,DO]
    const float* bias    = (const float*)d_in[4];   // [DO]

    float* out  = (float*)d_out;                       // [B,Q,DO]
    float* attn = out + (size_t)B * Q * DO;            // [B,Q,K]
    float* out1 = (float*)d_ws;                        // [B,Q,DV] scratch

    const float scale = 0.044194173824159216f;         // 1/sqrt(512)

    // 1) logits = scale * Q . K^T  -> attn region (f32)
    gemm_f16_wmma<false, false><<<dim3(KK / 128, Q / 128, B), 256, 0, stream>>>(
        queries, keys, nullptr, attn,
        /*Kd=*/D, /*lda=*/D, /*ldb=*/D, /*ldc=*/KK,
        (long)Q * D, (long)KK * D, (long)Q * KK, scale);

    // 2) sparsemax along K, in place
    sparsemax_rows<<<B * Q, 256, 0, stream>>>(attn);

    // 3) out1 = attn . V
    gemm_f16_wmma<true, false><<<dim3(DV / 128, Q / 128, B), 256, 0, stream>>>(
        attn, values, nullptr, out1,
        /*Kd=*/KK, /*lda=*/KK, /*ldb=*/DV, /*ldc=*/DV,
        (long)Q * KK, (long)KK * DV, (long)Q * DV, 1.0f);

    // 4) out = out1 . W + bias   (batch folded into M)
    gemm_f16_wmma<true, true><<<dim3(DO / 128, (B * Q) / 128, 1), 256, 0, stream>>>(
        out1, W, bias, out,
        /*Kd=*/DV, /*lda=*/DV, /*ldb=*/DO, /*ldc=*/DO,
        0L, 0L, 0L, 1.0f);
}
